// ModelInstructionAggregate_14989435863151
// MI455X (gfx1250) — compile-verified, matrix operand order checked by reading, not run on hardware
//
#include <hip/hip_runtime.h>
#include <hip/hip_bf16.h>
#include <stddef.h>

// ---------------------------------------------------------------------------
// MI455X / gfx1250 implementation.
//
// Phase 0: pack [Wih_t;Whh_t] -> bf16 [1024][512], Wih_i/Whh_i -> bf16
//          [1024][256], fuse biases (f32).
// Phase 1: token-level LSTM, 256 blocks x 32 instructions (two 16-row A
//          tiles -> each B fragment feeds 2 WMMAs, halving L2 weight
//          traffic vs 16-row blocks). Gates = [X_t|H] (32x512 bf16, LDS) x
//          Wstacked via v_wmma_f32_16x16x32_bf16, f32 accumulators seeded
//          with bias. C lives in VGPRs, H ping-pongs via LDS bf16.
// Phase 2: xg = emb @ Wih_i^T + b_i  as a plain WMMA GEMM -> f32 [8192][1024].
// Phase 3: sequential instruction LSTM: 1 persistent block; h packed into
//          row 0 of a WMMA A fragment (other rows zero) so each of the 8192
//          steps is 512 WMMAs instead of a VALU matvec; Whh stays L2-hot.
// Phase 4: out[n] = dot(outs[n], Wl) + bl.
// ---------------------------------------------------------------------------

typedef __attribute__((ext_vector_type(16))) __bf16 v16bf;
typedef __attribute__((ext_vector_type(8)))  __bf16 v8bf;
typedef __attribute__((ext_vector_type(8)))  float  v8f;

#define NI   8192
#define HID  256
#define TMAX 16
#define G4H  1024   // 4*HID
#define KST  512    // stacked K (E + H)
#define MB   32     // phase-1 row-block (2 WMMA M tiles)

static __device__ __forceinline__ __bf16 f2bf(float f) {
  unsigned u = __builtin_bit_cast(unsigned, f);
  unsigned r = (u + 0x7FFFu + ((u >> 16) & 1u)) >> 16;   // round-to-nearest-even
  unsigned short s = (unsigned short)r;
  return __builtin_bit_cast(__bf16, s);
}

static __device__ __forceinline__ v8f v8f_splat(float x) {
  v8f r;
#pragma unroll
  for (int i = 0; i < 8; i++) r[i] = x;
  return r;
}

static __device__ __forceinline__ v16bf bf16_zero() {
  v16bf r;
  __bf16 z = f2bf(0.0f);
#pragma unroll
  for (int i = 0; i < 16; i++) r[i] = z;
  return r;
}

// Load one 16x32 A-style bf16 fragment slice for this lane:
// elements 0..7  = p[0..7]   (K = base + 8*half + 0..7)
// elements 8..15 = p[16..23] (K = base + 16 + 8*half + 0..7)
static __device__ __forceinline__ v16bf load_frag(const __bf16* p) {
  v8bf lo = *(const v8bf*)(p);
  v8bf hi = *(const v8bf*)(p + 16);
  v16bf r;
#pragma unroll
  for (int e = 0; e < 8; e++) { r[e] = lo[e]; r[8 + e] = hi[e]; }
  return r;
}

static __device__ __forceinline__ float sigmoidf_(float x) {
  return 1.0f / (1.0f + __expf(-x));
}

static __device__ __forceinline__ v8f wmma_bf16(v16bf a, v16bf b, v8f c) {
  return __builtin_amdgcn_wmma_f32_16x16x32_bf16(
      /*neg_a=*/false, a, /*neg_b=*/false, b,
      /*c_mod=*/(short)0, c, /*reuse_a=*/false, /*reuse_b=*/false);
}

// ---------------------------------------------------------------------------
// Phase 0: weight packing
// ---------------------------------------------------------------------------
__global__ __launch_bounds__(256) void prep_kernel(
    const float* __restrict__ Wih_t, const float* __restrict__ Whh_t,
    const float* __restrict__ bih_t, const float* __restrict__ bhh_t,
    const float* __restrict__ Wih_i, const float* __restrict__ Whh_i,
    const float* __restrict__ bih_i, const float* __restrict__ bhh_i,
    __bf16* __restrict__ Wt, __bf16* __restrict__ Wi,
    __bf16* __restrict__ Whh, float* __restrict__ bt, float* __restrict__ bi) {
  const int col = blockIdx.x;     // 0..1023 (gate output column)
  const int k   = threadIdx.x;    // 0..255
  Wt[(size_t)col * KST + k]        = f2bf(Wih_t[(size_t)col * HID + k]);
  Wt[(size_t)col * KST + HID + k]  = f2bf(Whh_t[(size_t)col * HID + k]);
  Wi[(size_t)col * HID + k]        = f2bf(Wih_i[(size_t)col * HID + k]);
  Whh[(size_t)col * HID + k]       = f2bf(Whh_i[(size_t)col * HID + k]);
  if (k == 0) {
    bt[col] = bih_t[col] + bhh_t[col];
    bi[col] = bih_i[col] + bhh_i[col];
  }
}

// ---------------------------------------------------------------------------
// Phase 1: token-level LSTM, one block per 32 instructions (2 M tiles)
// ---------------------------------------------------------------------------
__global__ __launch_bounds__(256) void tok_lstm_kernel(
    const float* __restrict__ tokens,   // [NI][TMAX][HID]
    const int*   __restrict__ lengths,  // [NI]
    const __bf16* __restrict__ Wt,      // [G4H][KST] bf16, row = out col
    const float* __restrict__ bt,       // [G4H]
    __bf16* __restrict__ emb) {         // [NI][HID] bf16 out
  __shared__ __bf16 Abuf[MB][KST];      // [row][k]; k<256 = X_t, k>=256 = H
  __shared__ int lenS[MB];

  const int tid   = threadIdx.x;
  const int lane  = tid & 31;
  const int wv    = tid >> 5;           // wave 0..7 -> hidden cols [32w,32w+32)
  const int half  = lane >> 4;
  const int r     = lane & 15;
  const int ibase = blockIdx.x * MB;

  if (tid < MB) lenS[tid] = lengths[ibase + tid];
  for (int idx = tid; idx < MB * HID; idx += 256) {
    int row = idx >> 8, k = idx & 255;
    Abuf[row][HID + k] = f2bf(0.0f);    // H starts at zero
  }
  __syncthreads();

  int lenv[2][8];
#pragma unroll
  for (int m = 0; m < 2; m++)
#pragma unroll
    for (int v = 0; v < 8; v++) lenv[m][v] = lenS[m * 16 + v + 8 * half];

  float bias[2][4];
#pragma unroll
  for (int j = 0; j < 2; j++)
#pragma unroll
    for (int g = 0; g < 4; g++)
      bias[j][g] = bt[g * HID + wv * 32 + j * 16 + r];

  v8f creg[2][2];
#pragma unroll
  for (int m = 0; m < 2; m++)
#pragma unroll
    for (int j = 0; j < 2; j++) creg[m][j] = v8f_splat(0.0f);

  for (int t = 0; t < TMAX; t++) {
    {  // fill X_t region of A (each thread: one 32-element chunk of one row)
      int row = tid >> 3;               // 0..31
      int kc  = (tid & 7) * 32;
      const float* src =
          tokens + ((size_t)(ibase + row) * TMAX + t) * HID + kc;
#pragma unroll
      for (int q = 0; q < 32; q++) Abuf[row][kc + q] = f2bf(src[q]);
    }
    __syncthreads();

    v8f acc[2][2][4];
#pragma unroll
    for (int m = 0; m < 2; m++)
#pragma unroll
      for (int j = 0; j < 2; j++)
#pragma unroll
        for (int g = 0; g < 4; g++) acc[m][j][g] = v8f_splat(bias[j][g]);

    for (int kk = 0; kk < KST / 32; kk++) {           // 16 K-slices
      v16bf afrag0 = load_frag(&Abuf[r][kk * 32 + 8 * half]);
      v16bf afrag1 = load_frag(&Abuf[16 + r][kk * 32 + 8 * half]);
#pragma unroll
      for (int j = 0; j < 2; j++) {
#pragma unroll
        for (int g = 0; g < 4; g++) {
          int col = g * HID + wv * 32 + j * 16 + r;
          v16bf bfrag =
              load_frag(Wt + (size_t)col * KST + kk * 32 + 8 * half);
          acc[0][j][g] = wmma_bf16(afrag0, bfrag, acc[0][j][g]);  // B reused
          acc[1][j][g] = wmma_bf16(afrag1, bfrag, acc[1][j][g]);
        }
      }
    }
    __syncthreads();  // everyone done reading Abuf H region

#pragma unroll
    for (int m = 0; m < 2; m++) {
#pragma unroll
      for (int j = 0; j < 2; j++) {
#pragma unroll
        for (int v = 0; v < 8; v++) {
          if (t < lenv[m][v]) {         // mask: freeze state past length
            float iv = sigmoidf_(acc[m][j][0][v]);
            float fv = sigmoidf_(acc[m][j][1][v]);
            float gv = tanhf(acc[m][j][2][v]);
            float ov = sigmoidf_(acc[m][j][3][v]);
            float c2 = fv * creg[m][j][v] + iv * gv;
            creg[m][j][v] = c2;
            float h2 = ov * tanhf(c2);
            Abuf[m * 16 + v + 8 * half][HID + wv * 32 + j * 16 + r] =
                f2bf(h2);
          }
        }
      }
    }
    __syncthreads();
  }

  for (int idx = tid; idx < MB * HID; idx += 256) {
    int row = idx >> 8, k = idx & 255;
    emb[(size_t)(ibase + row) * HID + k] = Abuf[row][HID + k];
  }
}

// ---------------------------------------------------------------------------
// Phase 2: xg = emb @ Wih_i^T + b_i   (WMMA GEMM, f32 out)
// ---------------------------------------------------------------------------
__global__ __launch_bounds__(256) void ins_gemm_kernel(
    const __bf16* __restrict__ emb,   // [NI][HID]
    const __bf16* __restrict__ Wi,    // [G4H][HID]
    const float*  __restrict__ bi,    // [G4H]
    float* __restrict__ xg) {         // [NI][G4H]
  const int tid   = threadIdx.x;
  const int lane  = tid & 31;
  const int wv    = tid >> 5;         // wave handles cols [wv*128, wv*128+128)
  const int half  = lane >> 4;
  const int r     = lane & 15;
  const int ibase = blockIdx.x * 16;

  v8f acc[8];
#pragma unroll
  for (int tI = 0; tI < 8; tI++)
    acc[tI] = v8f_splat(bi[wv * 128 + tI * 16 + r]);

  for (int kk = 0; kk < HID / 32; kk++) {
    v16bf afrag =
        load_frag(emb + (size_t)(ibase + r) * HID + kk * 32 + 8 * half);
#pragma unroll
    for (int tI = 0; tI < 8; tI++) {
      int col = wv * 128 + tI * 16 + r;
      v16bf bfrag = load_frag(Wi + (size_t)col * HID + kk * 32 + 8 * half);
      acc[tI] = wmma_bf16(afrag, bfrag, acc[tI]);
    }
  }

#pragma unroll
  for (int tI = 0; tI < 8; tI++) {
#pragma unroll
    for (int v = 0; v < 8; v++) {
      xg[(size_t)(ibase + v + 8 * half) * G4H + wv * 128 + tI * 16 + r] =
          acc[tI][v];
    }
  }
}

// ---------------------------------------------------------------------------
// Phase 3: sequential instruction LSTM, single persistent block.
// h occupies row 0 of the WMMA A fragment; rows 1..15 are zero.
// ---------------------------------------------------------------------------
__global__ __launch_bounds__(256) void ins_lstm_kernel(
    const __bf16* __restrict__ Whh,   // [G4H][HID] bf16
    const float*  __restrict__ xg,    // [NI][G4H] (already includes b_i)
    float* __restrict__ outs) {       // [NI][HID]
  __shared__ __bf16 hbuf[HID];

  const int tid  = threadIdx.x;
  const int lane = tid & 31;
  const int wv   = tid >> 5;          // wave -> hidden cols [32w, 32w+32)
  const int half = lane >> 4;
  const int r    = lane & 15;

  for (int i = tid; i < HID; i += 256) hbuf[i] = f2bf(0.0f);
  float cst[2] = {0.0f, 0.0f};
  __syncthreads();

  for (int n = 0; n < NI; n++) {
    const float* xrow = xg + (size_t)n * G4H;
    v8f acc[2][4];
#pragma unroll
    for (int j = 0; j < 2; j++) {
#pragma unroll
      for (int g = 0; g < 4; g++) {
        acc[j][g] = v8f_splat(0.0f);
        // seed only the valid (M=0) position: lanes 0..15, element 0
        acc[j][g][0] =
            (half == 0) ? xrow[g * HID + (2 * wv + j) * 16 + r] : 0.0f;
      }
    }

    for (int kk = 0; kk < HID / 32; kk++) {
      v16bf afrag = bf16_zero();
      if (r == 0) {  // lanes 0 and 16 carry row M=0 of A
        afrag = load_frag(&hbuf[kk * 32 + 8 * half]);
      }
#pragma unroll
      for (int j = 0; j < 2; j++) {
#pragma unroll
        for (int g = 0; g < 4; g++) {
          int col = g * HID + (2 * wv + j) * 16 + r;
          v16bf bfrag =
              load_frag(Whh + (size_t)col * HID + kk * 32 + 8 * half);
          acc[j][g] = wmma_bf16(afrag, bfrag, acc[j][g]);
        }
      }
    }
    __syncthreads();  // all waves done reading hbuf

#pragma unroll
    for (int j = 0; j < 2; j++) {
      if (half == 0) {  // element 0 of lanes 0..15 holds row M=0 of D
        float iv = sigmoidf_(acc[j][0][0]);
        float fv = sigmoidf_(acc[j][1][0]);
        float gv = tanhf(acc[j][2][0]);
        float ov = sigmoidf_(acc[j][3][0]);
        float c2 = fv * cst[j] + iv * gv;
        cst[j] = c2;
        float h2 = ov * tanhf(c2);
        int col = (2 * wv + j) * 16 + r;
        hbuf[col] = f2bf(h2);
        outs[(size_t)n * HID + col] = h2;
      }
    }
    __syncthreads();
  }
}

// ---------------------------------------------------------------------------
// Phase 4: out[n] = dot(outs[n], Wl) + bl
// ---------------------------------------------------------------------------
__global__ __launch_bounds__(256) void head_kernel(
    const float* __restrict__ outs, const float* __restrict__ Wl,
    const float* __restrict__ bl, float* __restrict__ out) {
  int n = blockIdx.x * blockDim.x + threadIdx.x;
  if (n >= NI) return;
  float s = bl[0];
  const float* row = outs + (size_t)n * HID;
#pragma unroll 8
  for (int k = 0; k < HID; k++) s += row[k] * Wl[k];
  out[n] = s;
}

// ---------------------------------------------------------------------------
// Workspace layout (bytes); total ~46 MB
// ---------------------------------------------------------------------------
static const size_t O_WT   = 0;                                  // 1 MB
static const size_t O_WI   = O_WT + (size_t)G4H * KST * 2;       // 512 KB
static const size_t O_WHH  = O_WI + (size_t)G4H * HID * 2;       // 512 KB
static const size_t O_BT   = O_WHH + (size_t)G4H * HID * 2;      // 4 KB
static const size_t O_BI   = O_BT + (size_t)G4H * 4;             // 4 KB
static const size_t O_EMB  = O_BI + (size_t)G4H * 4;             // 4 MB
static const size_t O_XG   = O_EMB + (size_t)NI * HID * 2;       // 32 MB
static const size_t O_OUTS = O_XG + (size_t)NI * G4H * 4;        // 8 MB

extern "C" void kernel_launch(void* const* d_in, const int* in_sizes, int n_in,
                              void* d_out, int out_size, void* d_ws,
                              size_t ws_size, hipStream_t stream) {
  (void)in_sizes; (void)n_in; (void)out_size; (void)ws_size;
  const float* tokens = (const float*)d_in[0];
  const int*   lengths = (const int*)d_in[1];
  const float* Wih_t = (const float*)d_in[2];
  const float* Whh_t = (const float*)d_in[3];
  const float* bih_t = (const float*)d_in[4];
  const float* bhh_t = (const float*)d_in[5];
  const float* Wih_i = (const float*)d_in[6];
  const float* Whh_i = (const float*)d_in[7];
  const float* bih_i = (const float*)d_in[8];
  const float* bhh_i = (const float*)d_in[9];
  const float* Wl = (const float*)d_in[10];
  const float* bl = (const float*)d_in[11];

  char* ws = (char*)d_ws;
  __bf16* Wt   = (__bf16*)(ws + O_WT);
  __bf16* Wi   = (__bf16*)(ws + O_WI);
  __bf16* Whh  = (__bf16*)(ws + O_WHH);
  float*  bt   = (float*)(ws + O_BT);
  float*  bi   = (float*)(ws + O_BI);
  __bf16* emb  = (__bf16*)(ws + O_EMB);
  float*  xg   = (float*)(ws + O_XG);
  float*  outs = (float*)(ws + O_OUTS);
  float*  out  = (float*)d_out;

  prep_kernel<<<G4H, 256, 0, stream>>>(Wih_t, Whh_t, bih_t, bhh_t, Wih_i,
                                       Whh_i, bih_i, bhh_i, Wt, Wi, Whh, bt,
                                       bi);
  tok_lstm_kernel<<<NI / MB, 256, 0, stream>>>(tokens, lengths, Wt, bt, emb);
  ins_gemm_kernel<<<NI / 16, 256, 0, stream>>>(emb, Wi, bi, xg);
  ins_lstm_kernel<<<1, 256, 0, stream>>>(Whh, xg, outs);
  head_kernel<<<NI / 256, 256, 0, stream>>>(outs, Wl, bl, out);
}
